// SDT_55980603736284
// MI455X (gfx1250) — matrix operand order repaired and checked by reading, not compiled, not run on hardware
//
#include <hip/hip_runtime.h>
#include <hip/hip_bf16.h>

// Soft decision tree forward for MI455X (gfx1250, wave32, WMMA).
// Only the 6 chain nodes {0,1,3,7,15,31} contribute to the output, so we
// compute exactly those. Main GEMM runs on v_wmma_f32_16x16x32_bf16 with
// fp32 accumulation. x is converted fp32->bf16 in registers (no extra HBM
// traffic). The 6 W1 slices are pre-transposed to bf16 [node][h][k] in ws,
// then staged once per block into (bank-conflict-padded) LDS so B-fragment
// fetches are short-latency ds_load_b128 shared by all 8 waves instead of
// per-wave L2 traffic.

typedef __attribute__((ext_vector_type(16))) __bf16 v16bf;
typedef __attribute__((ext_vector_type(8)))  float  v8f;

#define SDT_DEPTH   6
#define SDT_HIDDEN  64
#define SDT_INDIM   128
#define SDT_BATCH   65536
#define ROW_BF16    136          // 128 + 8 pad -> 272B stride = 68 dwords:
                                 // lane l16 starts at bank 4*l16 (mod 64),
                                 // conflict-free b128 reads per half-wave
#define W1_LDS_BYTES (SDT_DEPTH * SDT_HIDDEN * ROW_BF16 * 2)   // 104448 B

// ---------------------------------------------------------------------------
// Prep: W1t[n6][h][k] = bf16( W1[(2^n6)-1][k][h] ),  n6 = 0..5  (row-major k)
// ---------------------------------------------------------------------------
__global__ void sdt_prep_kernel(const float* __restrict__ W1,
                                __bf16* __restrict__ W1t) {
    int idx = blockIdx.x * blockDim.x + threadIdx.x;  // 6*64*128 = 49152
    if (idx >= SDT_DEPTH * SDT_HIDDEN * SDT_INDIM) return;
    int n6  = idx / (SDT_HIDDEN * SDT_INDIM);
    int rem = idx - n6 * (SDT_HIDDEN * SDT_INDIM);
    int h   = rem / SDT_INDIM;
    int k   = rem - h * SDT_INDIM;
    int nid = (1 << n6) - 1;                           // chain node id
    W1t[idx] = (__bf16)W1[((size_t)nid * SDT_INDIM + k) * SDT_HIDDEN + h];
}

// ---------------------------------------------------------------------------
// Main: one wave = one 16-row tile of x. 6 nodes * 4 N-tiles * 4 K-steps
// = 96 WMMAs per wave; B fragments come from padded LDS.
// ---------------------------------------------------------------------------
__global__ void __launch_bounds__(256)
sdt_main_kernel(const float*  __restrict__ x,
                const __bf16* __restrict__ W1t,
                const float*  __restrict__ b1,
                const float*  __restrict__ W2,
                const float*  __restrict__ b2,
                const float*  __restrict__ leaf,
                float*        __restrict__ out) {
    extern __shared__ char w1_dyn[];            // padded weight tile (104448 B)
    __shared__ float leaf_s[64];
    __shared__ float p_lds[8][16][SDT_DEPTH];   // [wave][row in tile][node]

    const int tid = threadIdx.x;

    // ---- stage W1t -> LDS with 272B row stride (one time per block) ----
    // 6*64 rows * 128 bf16 = 6144 uint4 total; 24 per thread, coalesced.
    {
        const uint4* g = (const uint4*)W1t;
#pragma unroll
        for (int i = tid; i < (SDT_DEPTH * SDT_HIDDEN * SDT_INDIM * 2) / 16;
             i += 256) {
            int row = i >> 4;                   // 16 uint4 per 256B source row
            int col = i & 15;
            *(uint4*)(w1_dyn + (size_t)row * (ROW_BF16 * 2) + col * 16) = g[i];
        }
    }
    if (tid < 64) leaf_s[tid] = leaf[tid];
    __syncthreads();

    const int lane = tid & 31;
    const int w    = tid >> 5;          // wave in block (0..7)
    const int half = lane >> 4;         // 0: lanes 0-15, 1: lanes 16-31
    const int l16  = lane & 15;
    const int m0   = (blockIdx.x * 8 + w) * 16;   // first row of this tile

    // ---- A fragments (bf16 16x32, ISA layout), shared by all 6 nodes ----
    // lane: row = m0 + l16; elems 0-7 -> K = ks*32 + half*8 + i
    //                       elems 8-15 -> K = ks*32 + 16 + half*8 + i
    v16bf a[4];
    const float* xr = x + (size_t)(m0 + l16) * SDT_INDIM;
#pragma unroll
    for (int ks = 0; ks < 4; ++ks) {
        const float4* p0 = (const float4*)(xr + ks * 32 + half * 8);
        const float4* p1 = (const float4*)(xr + ks * 32 + 16 + half * 8);
        float4 f0 = p0[0], f1 = p0[1];
        float4 f2 = p1[0], f3 = p1[1];
        v16bf av;
        av[0]  = (__bf16)f0.x; av[1]  = (__bf16)f0.y;
        av[2]  = (__bf16)f0.z; av[3]  = (__bf16)f0.w;
        av[4]  = (__bf16)f1.x; av[5]  = (__bf16)f1.y;
        av[6]  = (__bf16)f1.z; av[7]  = (__bf16)f1.w;
        av[8]  = (__bf16)f2.x; av[9]  = (__bf16)f2.y;
        av[10] = (__bf16)f2.z; av[11] = (__bf16)f2.w;
        av[12] = (__bf16)f3.x; av[13] = (__bf16)f3.y;
        av[14] = (__bf16)f3.z; av[15] = (__bf16)f3.w;
        a[ks] = av;
    }

#pragma unroll
    for (int n = 0; n < SDT_DEPTH; ++n) {
        const int nid = (1 << n) - 1;   // original node index for b1/W2/b2
        float acc[8];
#pragma unroll
        for (int r = 0; r < 8; ++r) acc[r] = 0.0f;

#pragma unroll
        for (int t = 0; t < 4; ++t) {   // N-tiles: h = t*16 + N
            v8f c = {};
#pragma unroll
            for (int ks = 0; ks < 4; ++ks) {
                // B fragment from LDS: elems 0-7 -> K half*8 + i, elems 8-15
                // -> K 16 + half*8 + i of this 32-K step; N = l16. Two 16B ds loads.
                const char* bb = w1_dyn
                    + (size_t)(n * SDT_HIDDEN + t * 16 + l16) * (ROW_BF16 * 2)
                    + ks * 64 + half * 16;
                union { uint4 u[2]; v16bf v; } bf;
                bf.u[0] = *(const uint4*)bb;
                bf.u[1] = *(const uint4*)(bb + 32);
                c = __builtin_amdgcn_wmma_f32_16x16x32_bf16(
                        false, a[ks], false, bf.v, (short)0, c, false, false);
            }
            // C layout: c[r] = element (M = r + 8*half, N = l16)
            const float b1v = b1[nid * SDT_HIDDEN + t * 16 + l16];
            const float w2v = W2[nid * SDT_HIDDEN + t * 16 + l16];
#pragma unroll
            for (int r = 0; r < 8; ++r) {
                float hv = c[r] + b1v;
                hv = hv > 0.0f ? hv : 0.0f;       // ReLU
                acc[r] += hv * w2v;               // partial W2 dot over N
            }
        }
        // reduce over the 16 lanes of each half (xor 1,2,4,8 stays in-half)
#pragma unroll
        for (int r = 0; r < 8; ++r) {
#pragma unroll
            for (int off = 1; off < 16; off <<= 1)
                acc[r] += __shfl_xor(acc[r], off, 32);
        }
        const float b2v = b2[nid];
        if (l16 == 0) {
#pragma unroll
            for (int r = 0; r < 8; ++r) {
                float s  = acc[r] + b2v;
                float pv = 1.0f / (1.0f + __expf(-s));   // sigmoid
                p_lds[w][half * 8 + r][n] = pv;
            }
        }
    }

    // ---- leaf fold: lane m (<16) handles row m0+m; 6-level pairwise fold ----
    if (lane < 16) {
        float pr[SDT_DEPTH];
#pragma unroll
        for (int k = 0; k < SDT_DEPTH; ++k) pr[k] = p_lds[w][lane][k];

        float t[32];
#pragma unroll
        for (int j = 0; j < 32; ++j)
            t[j] = (1.0f - pr[0]) * leaf_s[2 * j] + pr[0] * leaf_s[2 * j + 1];
#pragma unroll
        for (int k = 1; k < SDT_DEPTH; ++k) {
#pragma unroll
            for (int j = 0; j < (32 >> k); ++j)
                t[j] = (1.0f - pr[k]) * t[2 * j] + pr[k] * t[2 * j + 1];
        }
        out[m0 + lane] = t[0];
    }
}

// ---------------------------------------------------------------------------
extern "C" void kernel_launch(void* const* d_in, const int* in_sizes, int n_in,
                              void* d_out, int out_size, void* d_ws, size_t ws_size,
                              hipStream_t stream) {
    const float* x    = (const float*)d_in[0];
    const float* W1   = (const float*)d_in[1];
    const float* b1   = (const float*)d_in[2];
    const float* W2   = (const float*)d_in[3];
    const float* b2   = (const float*)d_in[4];
    const float* leaf = (const float*)d_in[5];
    float*       out  = (float*)d_out;
    __bf16*      W1t  = (__bf16*)d_ws;          // 6*64*128*2 = 96 KB scratch

    const int prep_elems = SDT_DEPTH * SDT_HIDDEN * SDT_INDIM;
    sdt_prep_kernel<<<(prep_elems + 255) / 256, 256, 0, stream>>>(W1, W1t);

    // 256 threads = 8 waves = 128 rows per block; dynamic LDS holds the
    // padded bf16 weight tile (104448 B, within gfx1250's 320KB/WG limit).
    sdt_main_kernel<<<SDT_BATCH / 128, 256, W1_LDS_BYTES, stream>>>(
        x, W1t, b1, W2, b2, leaf, out);
}
